// TreeNN_65249143161598
// MI455X (gfx1250) — matrix operand order repaired and checked by reading, not compiled, not run on hardware
//
#include <hip/hip_runtime.h>

// ---------------------------------------------------------------------------
// TreeNN fused kernel for MI455X (gfx1250, wave32, WMMA)
//   features = relu(x @ Wf + bf)            [B,H]   bf16 WMMA, f32 accum
//   logits   = x @ Wr + br                  [B,L]   (softmax dropped: argmax-invariant)
//   choice   = argmax(logits)               [B]
//   out      = einsum(features, Wl[choice]) + bl[choice]   [B,C]
//
// v3: in-loop barriers wait on DScnt only (hand-rolled s_wait_dscnt +
//     s_barrier_signal/wait) so the pipelined global B/x loads stay in
//     flight across the LDS double-buffer barriers; __syncthreads() was
//     draining LOADcnt to 0 every K-step.
// ---------------------------------------------------------------------------

typedef __attribute__((ext_vector_type(16))) __bf16 v16bf;
typedef __attribute__((ext_vector_type(8)))  float  v8f;

#define BATCH   65536
#define DIN     784
#define HID     256
#define NLEAF   64
#define NCLS    10
#define KTILES  25          // K padded 784 -> 800 (25 x 32)
#define NFT     16          // feature n-tiles (256/16)
#define NRT     4           // router  n-tiles (64/16)
#define MROWS   64          // rows per workgroup

// Workgroup barrier that only waits for outstanding LDS ops (DScnt), so
// global loads issued before it remain in flight across the barrier.
__device__ __forceinline__ void wg_barrier_ds() {
    asm volatile("s_wait_dscnt 0x0\n\t"
                 "s_barrier_signal -1\n\t"
                 "s_barrier_wait -1" ::: "memory");
}

// ---------------------------------------------------------------------------
// Pack Wf (784x256) and Wr (784x64) into bf16 WMMA B-operand tiles.
//   lane n (n<16)  : column N = n,  16 bf16 = K 0..15 of the 32-K tile
//   lane n+16      : column N = n,  16 bf16 = K 16..31
// Tile = 1024 B contiguous; tiles ordered [ntile][ktile] for streaming.
// ---------------------------------------------------------------------------
__global__ __launch_bounds__(256) void prep_weights(const float* __restrict__ Wf,
                                                    const float* __restrict__ Wr,
                                                    __bf16* __restrict__ packed) {
    int tid = blockIdx.x * 256 + threadIdx.x;          // one bf16 element each
    const int total = (NFT + NRT) * KTILES * 512;
    if (tid >= total) return;
    int tile = tid >> 9;        // 512 elems per tile
    int e    = tid & 511;
    int lane = e >> 4;
    int j    = e & 15;
    float v = 0.0f;
    if (tile < NFT * KTILES) {
        int nt = tile / KTILES, kt = tile % KTILES;
        int col = nt * 16 + (lane & 15);
        int K   = kt * 32 + ((lane >> 4) << 4) + j;
        if (K < DIN) v = Wf[K * HID + col];
    } else {
        int t2 = tile - NFT * KTILES;
        int nt = t2 / KTILES, kt = t2 % KTILES;
        int col = nt * 16 + (lane & 15);
        int K   = kt * 32 + ((lane >> 4) << 4) + j;
        if (K < DIN) v = Wr[K * NLEAF + col];
    }
    packed[tid] = (__bf16)v;
}

// ---------------------------------------------------------------------------
// Main fused kernel: 256 threads = 8 waves handle 64 rows of x.
//   wave w: row pair base mtp = w>>2 (rows mtp*32 .. +32, two 16-row A tiles)
//           column group ng = w&3 : feature cols [ng*64, ng*64+64) + router
//           cols [ng*16, ng*16+16).  10 WMMA per K-step per wave.
// ---------------------------------------------------------------------------
__global__ __launch_bounds__(256) void treenn_main(
    const float* __restrict__ x,
    const float* __restrict__ bf_g,
    const float* __restrict__ br_g,
    const float* __restrict__ Wl,
    const float* __restrict__ bl,
    const __bf16* __restrict__ packed,
    float* __restrict__ out) {

    __shared__ __align__(16) __bf16 xt[2][MROWS][40];  // 64 rows x 32 K bf16, stride 80B
    __shared__ float logitsS[MROWS][68];
    __shared__ float featS[MROWS][260];
    __shared__ int   choiceS[MROWS];
    __shared__ float outS[MROWS][NCLS];

    const int t    = threadIdx.x;
    const int wave = t >> 5;
    const int lane = t & 31;
    const int lc   = lane & 15;
    const int m0   = blockIdx.x * MROWS;
    const int mtp  = wave >> 2;      // 0/1 -> rows mtp*32..mtp*32+31
    const int ng   = wave & 3;       // column group

    for (int i = t; i < MROWS * NCLS; i += 256) ((float*)outS)[i] = 0.0f;

    float biasF[4];
#pragma unroll
    for (int i = 0; i < 4; ++i) biasF[i] = bf_g[ng * 64 + i * 16 + lc];
    float biasR = br_g[ng * 16 + lc];

    // x-tile loader: thread -> (row 0..63, 8 consecutive K)
    const int lrow = t >> 2;
    const int lkq  = (t & 3) * 8;
    const float* xrow = x + (size_t)(m0 + lrow) * DIN;

    // B-operand stream bases (int4 units; 64 int4 per 1KB tile)
    const int4* pk = (const int4*)packed;
    int bofs[5];
#pragma unroll
    for (int i = 0; i < 4; ++i) bofs[i] = ((ng * 4 + i) * KTILES) * 64 + lane * 2;
    bofs[4] = (NFT * KTILES) * 64 + (ng * KTILES) * 64 + lane * 2;

    v8f zero = {0.f, 0.f, 0.f, 0.f, 0.f, 0.f, 0.f, 0.f};
    v8f accF[2][4]; v8f accR[2];
#pragma unroll
    for (int j = 0; j < 2; ++j) {
        accR[j] = zero;
#pragma unroll
        for (int i = 0; i < 4; ++i) accF[j][i] = zero;
    }

    // A fragment addressing in LDS (contiguous K: lane half h holds K 16h..16h+15, same as B pack)
    const int arow  = mtp * 32 + lc;
    const int akoff = (lane >> 4) * 16;

    union BFu { int4 i4[2]; v16bf v; };
    BFu Bb[2][5];                      // register double-buffer for B fragments

    auto loadB = [&](int ks, int slot) {
#pragma unroll
        for (int i = 0; i < 5; ++i) {
            const int4* p = pk + bofs[i] + ks * 64;
            Bb[slot][i].i4[0] = p[0];
            Bb[slot][i].i4[1] = p[1];
        }
    };
    auto loadx4 = [&](int ks, int h) -> float4 {
        int k = ks * 32 + lkq + h * 4;
        if (k + 3 < DIN) return *(const float4*)(xrow + k);
        float4 z; z.x = z.y = z.z = z.w = 0.0f; return z;
    };

    loadB(0, 0);
    float4 xr0 = loadx4(0, 0), xr1 = loadx4(0, 1);

#pragma unroll
    for (int ks = 0; ks < KTILES; ++ks) {
        const int buf = ks & 1;
        wg_barrier_ds();                        // buf consumers (ks-2) done
        {   // fp32 -> bf16, one 16-byte LDS store per thread
            union { __bf16 b[8]; int4 u; } p;
            p.b[0] = (__bf16)xr0.x; p.b[1] = (__bf16)xr0.y;
            p.b[2] = (__bf16)xr0.z; p.b[3] = (__bf16)xr0.w;
            p.b[4] = (__bf16)xr1.x; p.b[5] = (__bf16)xr1.y;
            p.b[6] = (__bf16)xr1.z; p.b[7] = (__bf16)xr1.w;
            *(int4*)&xt[buf][lrow][lkq] = p.u;
        }
        // Pipeline next x + next B: pure global loads, issued before the
        // barrier so they stay in flight across it (barrier waits DScnt only).
        if (ks + 1 < KTILES) {
            xr0 = loadx4(ks + 1, 0);
            xr1 = loadx4(ks + 1, 1);
            loadB(ks + 1, (ks + 1) & 1);
        }
        if (ks + 2 < KTILES)                    // global_prefetch_b8 on x stream
            __builtin_prefetch(xrow + (ks + 2) * 32 + lkq, 0, 1);
        wg_barrier_ds();                        // xt[buf] visible to all waves

        // Two A fragments from LDS (4x ds_load_b128)
        union { int4 i4[2]; v16bf v; } A0, A1;
        A0.i4[0] = *(const int4*)&xt[buf][arow][akoff];
        A0.i4[1] = *(const int4*)&xt[buf][arow][akoff + 8];
        A1.i4[0] = *(const int4*)&xt[buf][arow + 16][akoff];
        A1.i4[1] = *(const int4*)&xt[buf][arow + 16][akoff + 8];

        // 10 WMMAs reusing the 5 current B fragments
#pragma unroll
        for (int i = 0; i < 4; ++i) {
            accF[0][i] = __builtin_amdgcn_wmma_f32_16x16x32_bf16(
                false, A0.v, false, Bb[buf][i].v, (short)0, accF[0][i], false, false);
            accF[1][i] = __builtin_amdgcn_wmma_f32_16x16x32_bf16(
                false, A1.v, false, Bb[buf][i].v, (short)0, accF[1][i], false, false);
        }
        accR[0] = __builtin_amdgcn_wmma_f32_16x16x32_bf16(
            false, A0.v, false, Bb[buf][4].v, (short)0, accR[0], false, false);
        accR[1] = __builtin_amdgcn_wmma_f32_16x16x32_bf16(
            false, A1.v, false, Bb[buf][4].v, (short)0, accR[1], false, false);
    }

    // Epilogue: scatter C tiles to LDS.
    // C layout: VGPR g, lanes 0-15 -> (M=g, N=lane); lanes 16-31 -> (M=g+8, N=lane-16)
#pragma unroll
    for (int j = 0; j < 2; ++j) {
        const int rBase = mtp * 32 + j * 16 + ((lane >> 4) ? 8 : 0);
#pragma unroll
        for (int g = 0; g < 8; ++g)
            logitsS[rBase + g][ng * 16 + lc] = accR[j][g] + biasR;
#pragma unroll
        for (int i = 0; i < 4; ++i) {
#pragma unroll
            for (int g = 0; g < 8; ++g) {
                float f = accF[j][i][g] + biasF[i];
                featS[rBase + g][ng * 64 + i * 16 + lc] = f > 0.0f ? f : 0.0f;
            }
        }
    }
    __syncthreads();

    // Per-row argmax over 64 logits (strict '>' => first max, like jnp.argmax)
    if (t < MROWS) {
        float best = logitsS[t][0]; int bi = 0;
        for (int c = 1; c < NLEAF; ++c) {
            float v = logitsS[t][c];
            if (v > best) { best = v; bi = c; }
        }
        choiceS[t] = bi;
    }
    __syncthreads();

    // Leaf dispatch: 4 threads per row, each covers 64 of H=256
    {
        const int row = t & 63, sl = t >> 6;
        const int ch  = choiceS[row];
        const float* wl = Wl + (size_t)ch * (HID * NCLS) + (sl * 64) * NCLS;
        float p[NCLS];
#pragma unroll
        for (int c = 0; c < NCLS; ++c) p[c] = 0.0f;
        for (int h = 0; h < 64; ++h) {
            float f = featS[row][sl * 64 + h];
#pragma unroll
            for (int c = 0; c < NCLS; ++c) p[c] += f * wl[h * NCLS + c];
        }
#pragma unroll
        for (int c = 0; c < NCLS; ++c) atomicAdd(&outS[row][c], p[c]);  // ds_add_f32
    }
    __syncthreads();

    for (int i = t; i < MROWS * NCLS; i += 256) {
        int row = i / NCLS, c = i % NCLS;
        out[(size_t)(m0 + row) * NCLS + c] = outS[row][c] + bl[choiceS[row] * NCLS + c];
    }
}

// ---------------------------------------------------------------------------
extern "C" void kernel_launch(void* const* d_in, const int* in_sizes, int n_in,
                              void* d_out, int out_size, void* d_ws, size_t ws_size,
                              hipStream_t stream) {
    const float* x  = (const float*)d_in[0];
    const float* Wf = (const float*)d_in[1];
    const float* bf = (const float*)d_in[2];
    const float* Wr = (const float*)d_in[3];
    const float* br = (const float*)d_in[4];
    const float* Wl = (const float*)d_in[5];
    const float* bl = (const float*)d_in[6];
    float* out = (float*)d_out;
    __bf16* packed = (__bf16*)d_ws;   // needs (16+4)*25*1024 = 512000 bytes

    const int prepN = (NFT + NRT) * KTILES * 512;   // 256000 elements
    prep_weights<<<(prepN + 255) / 256, 256, 0, stream>>>(Wf, Wr, packed);
    treenn_main<<<BATCH / MROWS, 256, 0, stream>>>(x, bf, br, Wl, bl, packed, out);
}
